// DiscountedL1_807453852341
// MI455X (gfx1250) — compile-verified, matrix operand order checked by reading, not run on hardware
//
#include <hip/hip_runtime.h>
#include <math.h>

typedef __attribute__((ext_vector_type(2)))  float    v2f;
typedef __attribute__((ext_vector_type(8)))  float    v8f;
typedef __attribute__((ext_vector_type(16))) _Float16 v16h;

#define Bn   8
#define Cn   3
#define Hn   256
#define Wn   256
#define HWn  (Hn * Wn)          /* 65536  */
#define BHW  (Bn * HWn)         /* 524288 */
#define BCHW (Bn * Cn * HWn)    /* 1572864 */
#define BIGD 512.0f
#define LOG2_GAMMA (-0.014499569695115089f)  /* log2(0.99) */

#if defined(__has_builtin)
#if __has_builtin(__builtin_amdgcn_wmma_f32_16x16x4_f32)
#define USE_F32_WMMA 1
#endif
#endif

// ---------------- Stage 0: zero per-batch border flags ----------------
__global__ void k_init(int* __restrict__ flags) {
    if (threadIdx.x < Bn) flags[threadIdx.x] = 0;
}

// ---------------- Stage 1: border = maxpool3x3(mask) - mask > 0.5 ----------------
__global__ void k_border(const float* __restrict__ mask,
                         unsigned char* __restrict__ border,
                         int* __restrict__ flags) {
    int p = blockIdx.x * blockDim.x + threadIdx.x;
    if (p >= BHW) return;
    int b  = p >> 16;
    int hw = p & 65535;
    int h  = hw >> 8;
    int w  = hw & 255;
    const float* mb = mask + b * HWn;
    float self = mb[hw];
    float mp = self;
    #pragma unroll
    for (int dh = -1; dh <= 1; ++dh) {
        int hh = h + dh;
        if (hh < 0 || hh >= Hn) continue;
        #pragma unroll
        for (int dw = -1; dw <= 1; ++dw) {
            int ww = w + dw;
            if (ww < 0 || ww >= Wn) continue;
            mp = fmaxf(mp, mb[(hh << 8) + ww]);
        }
    }
    unsigned char isb = ((mp - self) > 0.5f) ? 1 : 0;
    border[p] = isb;
    if (isb) atomicOr(&flags[b], 1);
}

// ---------------- Stage 2: per-column vertical 1D distance, squared ----------------
__global__ void k_colscan(const unsigned char* __restrict__ border,
                          float* __restrict__ g2) {
    int t = blockIdx.x * blockDim.x + threadIdx.x;  // (b, w): adjacent threads = adjacent cols (coalesced)
    if (t >= Bn * Wn) return;
    int b = t >> 8;
    int w = t & 255;
    const unsigned char* bb = border + b * HWn + w;
    float* gg = g2 + b * HWn + w;
    float d = BIGD;
    for (int h = 0; h < Hn; ++h) {            // down scan: store raw distance
        d = bb[h << 8] ? 0.0f : d + 1.0f;
        gg[h << 8] = d;
    }
    float u = BIGD;
    for (int h = Hn - 1; h >= 0; --h) {       // up scan: combine, clamp, square
        u = bb[h << 8] ? 0.0f : u + 1.0f;
        float g = fminf(fminf(gg[h << 8], u), BIGD);
        gg[h << 8] = g * g;
    }
}

// ---------------- Stage 3: per-row min-plus + weight = gamma^D * mask (in-place) ----------------
__global__ void k_rowdist(float* __restrict__ g2w,
                          const float* __restrict__ mask,
                          const int* __restrict__ flags) {
    __shared__ float s[Wn];
    int row = blockIdx.x;          // b*Hn + h
    int t   = threadIdx.x;         // column
    int b   = row >> 8;
    float* rowp = g2w + row * Wn;
    s[t] = rowp[t];
    __syncthreads();
    float m = mask[row * Wn + t];
    float wgt;
    if (m > 0.5f) {
        if (flags[b]) {
            float best = 1e30f;
            for (int c = 0; c < Wn; ++c) {
                float dx = (float)(t - c);
                best = fminf(best, dx * dx + s[c]);
            }
            float D = sqrtf(best);
            wgt = exp2f(D * LOG2_GAMMA);   // 0.99^D
        } else {
            wgt = 1.0f;                    // no border anywhere -> dist = 0
        }
    } else {
        wgt = 0.0f;
    }
    rowp[t] = wgt;                         // overwrite g2 with weight
}

// ---------------- Stage 4: weighted L1 partial sums via WMMA ----------------
__global__ void k_wsum(const float* __restrict__ rec, const float* __restrict__ im,
                       const float* __restrict__ weight, float* __restrict__ partials) {
    __shared__ float sw[8];
    int lane        = threadIdx.x & 31;
    int waveInBlock = threadIdx.x >> 5;
    // Force the wave index into an SGPR: the grid-stride loop below becomes pure
    // scalar control flow (s_cbranch), guaranteeing EXEC == all-1s at every WMMA
    // (ISA requirement), instead of a compiler-generated EXEC-masked loop.
    int wave = __builtin_amdgcn_readfirstlane(blockIdx.x * 8 + waveInBlock);
    const int totalWaves = 64 * 8;                 // grid = 64 blocks of 256

    v8f acc = {};
    // BHW/64 = 8192 wave-iterations, 64 pixel-contributions per WMMA (16x16x4 f32)
    for (int it = wave; it < BHW / 64; it += totalWaves) {
        int p   = it * 64 + 2 * lane;      // even; a (p, p+1) pair never crosses a batch boundary
        int b   = p >> 16;
        int pix = p & 65535;
        int i0  = b * (Cn * HWn) + pix;
        v2f w  = *(const v2f*)(weight + p);
        v2f r0 = *(const v2f*)(rec + i0);
        v2f e0 = *(const v2f*)(im  + i0);
        v2f r1 = *(const v2f*)(rec + i0 + HWn);
        v2f e1 = *(const v2f*)(im  + i0 + HWn);
        v2f r2 = *(const v2f*)(rec + i0 + 2 * HWn);
        v2f e2 = *(const v2f*)(im  + i0 + 2 * HWn);
        v2f a;
        a[0] = w[0] * (fabsf(r0[0] - e0[0]) + fabsf(r1[0] - e1[0]) + fabsf(r2[0] - e2[0]));
        a[1] = w[1] * (fabsf(r0[1] - e0[1]) + fabsf(r1[1] - e1[1]) + fabsf(r2[1] - e2[1]));
#ifdef USE_F32_WMMA
        v2f ones; ones[0] = 1.0f; ones[1] = 1.0f;
        // D = A * ones + C : exact f32 row-sum accumulation on the matrix pipe
        acc = __builtin_amdgcn_wmma_f32_16x16x4_f32(false, a, false, ones,
                                                    (short)0, acc, false, false);
#else
        v16h ah = {};
        ah[0] = (_Float16)a[0];
        ah[1] = (_Float16)a[1];
        v16h ones;
        #pragma unroll
        for (int j = 0; j < 16; ++j) ones[j] = (_Float16)1.0f;
        acc = __builtin_amdgcn_wmma_f32_16x16x32_f16(false, ah, false, ones,
                                                     (short)0, acc, false, false);
#endif
    }
    // every column of D is the identical row-sum -> total is 16x; fixed in final scale
    float s = acc[0] + acc[1] + acc[2] + acc[3] + acc[4] + acc[5] + acc[6] + acc[7];
    #pragma unroll
    for (int off = 16; off > 0; off >>= 1) s += __shfl_down(s, off, 32);
    if (lane == 0) sw[waveInBlock] = s;
    __syncthreads();
    if (threadIdx.x == 0) {
        float t = 0.0f;
        for (int i = 0; i < 8; ++i) t += sw[i];
        partials[blockIdx.x] = t;
    }
}

// ---------------- Stage 5: final reduce + mean scale ----------------
__global__ void k_final(const float* __restrict__ partials, float* __restrict__ out) {
    __shared__ float s[64];
    int t = threadIdx.x;
    s[t] = partials[t];
    __syncthreads();
    for (int off = 32; off > 0; off >>= 1) {
        if (t < off) s[t] += s[t + off];
        __syncthreads();
    }
    if (t == 0) out[0] = s[0] * (1.0f / (16.0f * (float)BCHW));
}

extern "C" void kernel_launch(void* const* d_in, const int* in_sizes, int n_in,
                              void* d_out, int out_size, void* d_ws, size_t ws_size,
                              hipStream_t stream) {
    (void)in_sizes; (void)n_in; (void)out_size; (void)ws_size;
    const float* rec  = (const float*)d_in[0];
    const float* im   = (const float*)d_in[1];
    const float* mask = (const float*)d_in[2];
    float* out = (float*)d_out;

    char* ws = (char*)d_ws;
    int*           flags    = (int*)ws;                         //   32 B
    float*         partials = (float*)(ws + 256);               //  256 B
    unsigned char* border   = (unsigned char*)(ws + 1024);      //  BHW bytes
    float*         g2w      = (float*)(ws + 1024 + BHW);        //  BHW*4 bytes (offset 525312, 1 KB aligned)

    hipLaunchKernelGGL(k_init,    dim3(1),           dim3(64),  0, stream, flags);
    hipLaunchKernelGGL(k_border,  dim3(BHW / 256),   dim3(256), 0, stream, mask, border, flags);
    hipLaunchKernelGGL(k_colscan, dim3((Bn*Wn)/256), dim3(256), 0, stream, border, g2w);
    hipLaunchKernelGGL(k_rowdist, dim3(Bn * Hn),     dim3(Wn),  0, stream, g2w, mask, flags);
    hipLaunchKernelGGL(k_wsum,    dim3(64),          dim3(256), 0, stream, rec, im, g2w, partials);
    hipLaunchKernelGGL(k_final,   dim3(1),           dim3(64),  0, stream, partials, out);
}
